// FiDO_T5Attention_5600637354472
// MI455X (gfx1250) — matrix operand order repaired and checked by reading, not compiled, not run on hardware
//
#include <hip/hip_runtime.h>
#include <hip/hip_bf16.h>
#include <math.h>

// ---------------------------------------------------------------------------
// FiDO T5 attention (GQA-folded) for gfx1250 / MI455X.
// Pipeline: fold weights -> bias LUT -> QKV GEMM (bf16 WMMA) ->
//           flash attention (bf16 WMMA, online softmax) -> output GEMM.
// CDNA5 paths: v_wmma_f32_16x16x32_bf16, global_load_async_to_lds_b128
// (ASYNCcnt), ds_load_b128 fragment reads.
// ---------------------------------------------------------------------------

typedef __bf16 v16bf __attribute__((ext_vector_type(16)));
typedef __bf16 v4bf  __attribute__((ext_vector_type(4)));
typedef float  v8f   __attribute__((ext_vector_type(8)));

#define BATCH   4
#define SEQ     2048
#define DMODEL  1024
#define KVH     4
#define DHEAD   64
#define MTOT    (BATCH * SEQ)      // 8192 rows
#define QKVW    768                // 3 * KVH * DHEAD
#define OW      256                // KVH * DHEAD
#define PAD     8                  // LDS row pad (16B, keeps b128 alignment)

// ---- CDNA5 async global->LDS copy (16B per lane), tracked by ASYNCcnt -----
__device__ __forceinline__ void async_copy_b128(void* lds_dst, const void* gsrc) {
    unsigned off = (unsigned)(unsigned long long)lds_dst;   // LDS offset = addr[31:0]
    asm volatile("global_load_async_to_lds_b128 %0, %1, off"
                 :: "v"(off), "v"((unsigned long long)gsrc)
                 : "memory");
}
__device__ __forceinline__ void wait_async() {
    asm volatile("s_wait_asynccnt 0x0" ::: "memory");
}

// -------------------------------- weight folding ---------------------------
__global__ __launch_bounds__(256) void fold_kernel(
    const float* __restrict__ Wq, const float* __restrict__ Wk,
    const float* __restrict__ Wv, const float* __restrict__ Wo,
    __bf16* __restrict__ Wcat,    // [768, 1024] rows: 0..255 Q, 256..511 K, 512..767 V
    __bf16* __restrict__ WoEff)   // [1024, 256]
{
    int i = blockIdx.x * blockDim.x + threadIdx.x;
    const int total1 = QKVW * DMODEL;
    if (i < total1) {
        int row = i >> 10, col = i & 1023;
        int which = row >> 8;           // 0=Q 1=K 2=V
        int hd = row & 255;
        int h = hd >> 6, d = hd & 63;
        const float* W = (which == 0) ? Wq : (which == 1) ? Wk : Wv;
        float s = 0.f;
        #pragma unroll
        for (int g = 0; g < 4; ++g)
            s += W[(size_t)((g * 4 + h) * 64 + d) * DMODEL + col];
        if (which != 0) s *= 0.25f;     // K,V are group means
        Wcat[(size_t)row * DMODEL + col] = (__bf16)s;
    } else {
        int j = i - total1;
        if (j < DMODEL * OW) {
            int m = j >> 8, c = j & 255;
            float s = 0.f;
            #pragma unroll
            for (int g = 0; g < 4; ++g)
                s += Wo[(size_t)m * DMODEL + g * OW + c];
            WoEff[(size_t)m * OW + c] = (__bf16)s;
        }
    }
}

// ------------------------------ bias LUT -----------------------------------
__global__ __launch_bounds__(256) void bias_kernel(
    const float* __restrict__ rel_bias,  // [32, 16]
    float* __restrict__ biasTab)         // [4, 4096]
{
    int i = blockIdx.x * blockDim.x + threadIdx.x;
    if (i >= KVH * 4096) return;
    int h = i >> 12, idx = i & 4095;
    if (idx >= 2 * SEQ - 1) { biasTab[i] = 0.f; return; }
    int rp = idx - (SEQ - 1);            // memory_pos - context_pos
    int bucket = (rp > 0) ? 16 : 0;      // nb = 16
    int rpa = rp < 0 ? -rp : rp;
    if (rpa < 8) {
        bucket += rpa;                   // max_exact = 8
    } else {
        int large = 8 + (int)(logf((float)rpa * (1.0f / 8.0f)) *
                              (8.0f / logf(16.0f)));
        bucket += (large < 15) ? large : 15;
    }
    float s = 0.f;
    #pragma unroll
    for (int g = 0; g < 4; ++g)
        s += rel_bias[bucket * 16 + g * 4 + h];
    biasTab[i] = 0.25f * s;              // mean over groups
}

// ---------------- WMMA fragment helpers (contiguous k order, same for A and B) -----------
// A (16x32 bf16): lane = M row (lm), VGPR v -> K = 16*half + 2v, 2v+1
// B (32x16 bf16): lane = N col (lm), VGPR v -> K = 16*half + 2v, 2v+1
__device__ __forceinline__ v16bf load_afrag(const __bf16* rowp, int half) {
    v16bf a;
    #pragma unroll
    for (int v = 0; v < 8; ++v) {
        int kb = 16 * half + 2 * v;
        a[2 * v]     = rowp[kb];
        a[2 * v + 1] = rowp[kb + 1];
    }
    return a;
}

// ------------------------- GEMM 1: fp32 A, bf16 out ------------------------
// C[M,Nout] = A[M,K] * W[Nout,K]^T.  Block 128x64, 4 waves, wave: 32x64.
#define TM 128
#define TN 64
#define TK 32
__global__ __launch_bounds__(128) void gemm_f32a_kernel(
    const float* __restrict__ A, int K, int Nout,
    const __bf16* __restrict__ W, __bf16* __restrict__ C)
{
    __shared__ __bf16 As[TM][TK + PAD];
    __shared__ __bf16 Bs[TN][TK + PAD];

    const int bm = blockIdx.x, bn = blockIdx.y;
    const int tid = threadIdx.x;
    const int wave = tid >> 5, lane = tid & 31;
    const int half = lane >> 4, lm = lane & 15;

    v8f acc[2][4] = {};

    for (int k0 = 0; k0 < K; k0 += TK) {
        // A tile: 128x32 fp32 -> bf16. 1024 float4 chunks / 128 thr = 8 iters.
        #pragma unroll
        for (int it = 0; it < 8; ++it) {
            int i = tid + it * 128;
            int r = i >> 3, c4 = (i & 7) * 4;
            const float4 v = *reinterpret_cast<const float4*>(
                A + (size_t)(bm * TM + r) * K + k0 + c4);
            v4bf p = { (__bf16)v.x, (__bf16)v.y, (__bf16)v.z, (__bf16)v.w };
            *reinterpret_cast<v4bf*>(&As[r][c4]) = p;
        }
        // W tile: 64x32 bf16 via async copy. 256 x 16B / 128 thr = 2 iters.
        #pragma unroll
        for (int it = 0; it < 2; ++it) {
            int i = tid + it * 128;
            int r = i >> 2, c8 = (i & 3) * 8;
            async_copy_b128(&Bs[r][c8],
                            W + (size_t)(bn * TN + r) * K + k0 + c8);
        }
        wait_async();
        __syncthreads();

        v16bf a0 = load_afrag(&As[wave * 32 + lm][0], half);
        v16bf a1 = load_afrag(&As[wave * 32 + 16 + lm][0], half);
        #pragma unroll
        for (int j = 0; j < 4; ++j) {
            v16bf bfrag;
            int bcol = j * 16 + lm;
            #pragma unroll
            for (int v = 0; v < 8; ++v) {
                int kb = 16 * half + 2 * v;
                bfrag[2 * v]     = Bs[bcol][kb];
                bfrag[2 * v + 1] = Bs[bcol][kb + 1];
            }
            acc[0][j] = __builtin_amdgcn_wmma_f32_16x16x32_bf16(
                false, a0, false, bfrag, (short)0, acc[0][j], false, false);
            acc[1][j] = __builtin_amdgcn_wmma_f32_16x16x32_bf16(
                false, a1, false, bfrag, (short)0, acc[1][j], false, false);
        }
        __syncthreads();
    }

    #pragma unroll
    for (int u = 0; u < 2; ++u)
        #pragma unroll
        for (int j = 0; j < 4; ++j)
            #pragma unroll
            for (int r = 0; r < 8; ++r) {
                int row = bm * TM + wave * 32 + u * 16 + r + 8 * half;
                int col = bn * TN + j * 16 + lm;
                C[(size_t)row * Nout + col] = (__bf16)acc[u][j][r];
            }
}

// ------------------------- GEMM 2: bf16 A, fp32 out ------------------------
__global__ __launch_bounds__(128) void gemm_bf16a_kernel(
    const __bf16* __restrict__ A, int K, int Nout,
    const __bf16* __restrict__ W, float* __restrict__ C)
{
    __shared__ __bf16 As[TM][TK + PAD];
    __shared__ __bf16 Bs[TN][TK + PAD];

    const int bm = blockIdx.x, bn = blockIdx.y;
    const int tid = threadIdx.x;
    const int wave = tid >> 5, lane = tid & 31;
    const int half = lane >> 4, lm = lane & 15;

    v8f acc[2][4] = {};

    for (int k0 = 0; k0 < K; k0 += TK) {
        // A tile: 128x32 bf16 via async. 512 x 16B / 128 thr = 4 iters.
        #pragma unroll
        for (int it = 0; it < 4; ++it) {
            int i = tid + it * 128;
            int r = i >> 2, c8 = (i & 3) * 8;
            async_copy_b128(&As[r][c8],
                            A + (size_t)(bm * TM + r) * K + k0 + c8);
        }
        // W tile: 64x32 bf16 via async. 2 iters.
        #pragma unroll
        for (int it = 0; it < 2; ++it) {
            int i = tid + it * 128;
            int r = i >> 2, c8 = (i & 3) * 8;
            async_copy_b128(&Bs[r][c8],
                            W + (size_t)(bn * TN + r) * K + k0 + c8);
        }
        wait_async();
        __syncthreads();

        v16bf a0 = load_afrag(&As[wave * 32 + lm][0], half);
        v16bf a1 = load_afrag(&As[wave * 32 + 16 + lm][0], half);
        #pragma unroll
        for (int j = 0; j < 4; ++j) {
            v16bf bfrag;
            int bcol = j * 16 + lm;
            #pragma unroll
            for (int v = 0; v < 8; ++v) {
                int kb = 16 * half + 2 * v;
                bfrag[2 * v]     = Bs[bcol][kb];
                bfrag[2 * v + 1] = Bs[bcol][kb + 1];
            }
            acc[0][j] = __builtin_amdgcn_wmma_f32_16x16x32_bf16(
                false, a0, false, bfrag, (short)0, acc[0][j], false, false);
            acc[1][j] = __builtin_amdgcn_wmma_f32_16x16x32_bf16(
                false, a1, false, bfrag, (short)0, acc[1][j], false, false);
        }
        __syncthreads();
    }

    #pragma unroll
    for (int u = 0; u < 2; ++u)
        #pragma unroll
        for (int j = 0; j < 4; ++j)
            #pragma unroll
            for (int r = 0; r < 8; ++r) {
                int row = bm * TM + wave * 32 + u * 16 + r + 8 * half;
                int col = bn * TN + j * 16 + lm;
                C[(size_t)row * Nout + col] = acc[u][j][r];
            }
}

// ------------------------------ flash attention ----------------------------
// grid: (SEQ/64, BATCH*KVH). 4 waves/block; wave owns 16 query rows.
#define CHK 64
__global__ __launch_bounds__(128) void attn_kernel(
    const __bf16* __restrict__ QKV,     // [MTOT, 768] : Q|K|V
    const float*  __restrict__ biasTab, // [4, 4096]
    __bf16*       __restrict__ Obuf)    // [MTOT, 256]
{
    __shared__ __bf16 Kt[CHK][DHEAD + PAD];
    __shared__ __bf16 Vt[CHK][DHEAD + PAD];
    __shared__ __bf16 Pst[4][16][CHK + PAD];

    const int bh = blockIdx.y;
    const int b = bh >> 2, h = bh & 3;
    const int tid = threadIdx.x;
    const int wave = tid >> 5, lane = tid & 31;
    const int half = lane >> 4, lm = lane & 15;
    const int qbase = blockIdx.x * 64 + wave * 16;

    // Preload Q fragments (head dim 64 -> 2 wmma K-steps)
    v16bf aq[2];
    {
        const __bf16* Qp = QKV + ((size_t)(b * SEQ + qbase + lm)) * QKVW + h * DHEAD;
        aq[0] = load_afrag(Qp, half);
        aq[1] = load_afrag(Qp + 32, half);
    }

    float rowmax[8], rowsum[8];
    #pragma unroll
    for (int r = 0; r < 8; ++r) { rowmax[r] = -1e30f; rowsum[r] = 0.f; }
    v8f o[4] = {};

    for (int s0 = 0; s0 < SEQ; s0 += CHK) {
        // K/V chunk stage via async copies: 64x64 bf16 each.
        #pragma unroll
        for (int it = 0; it < 4; ++it) {
            int i = tid + it * 128;
            int r = i >> 3, c8 = (i & 7) * 8;
            size_t base = ((size_t)(b * SEQ + s0 + r)) * QKVW + h * DHEAD + c8;
            async_copy_b128(&Kt[r][c8], QKV + 256 + base);
            async_copy_b128(&Vt[r][c8], QKV + 512 + base);
        }
        wait_async();
        __syncthreads();

        // ---- scores S = Q Kt^T (16 x 64), fp32 accum
        v8f sc[4] = {};
        #pragma unroll
        for (int t = 0; t < 2; ++t) {
            #pragma unroll
            for (int j = 0; j < 4; ++j) {
                v16bf bk;                        // B[kd][key]
                int key = j * 16 + lm;
                #pragma unroll
                for (int v = 0; v < 8; ++v) {
                    int kb = 32 * t + 16 * half + 2 * v;
                    bk[2 * v]     = Kt[key][kb];
                    bk[2 * v + 1] = Kt[key][kb + 1];
                }
                sc[j] = __builtin_amdgcn_wmma_f32_16x16x32_bf16(
                    false, aq[t], false, bk, (short)0, sc[j], false, false);
            }
        }

        // ---- bias add
        #pragma unroll
        for (int j = 0; j < 4; ++j) {
            #pragma unroll
            for (int r = 0; r < 8; ++r) {
                int qn = qbase + r + 8 * half;
                int sp = s0 + j * 16 + lm;
                sc[j][r] += biasTab[h * 4096 + (sp - qn + (SEQ - 1))];
            }
        }

        // ---- online softmax (row lives in a 16-lane half)
        #pragma unroll
        for (int r = 0; r < 8; ++r) {
            float tm = fmaxf(fmaxf(sc[0][r], sc[1][r]), fmaxf(sc[2][r], sc[3][r]));
            #pragma unroll
            for (int m = 1; m < 16; m <<= 1)
                tm = fmaxf(tm, __shfl_xor(tm, m, 32));
            float nm = fmaxf(rowmax[r], tm);
            float scale = __expf(rowmax[r] - nm);
            rowmax[r] = nm;
            float ps = 0.f;
            #pragma unroll
            for (int j = 0; j < 4; ++j) {
                float p = __expf(sc[j][r] - nm);
                sc[j][r] = p;
                ps += p;
            }
            #pragma unroll
            for (int m = 1; m < 16; m <<= 1)
                ps += __shfl_xor(ps, m, 32);
            rowsum[r] = rowsum[r] * scale + ps;
            #pragma unroll
            for (int j = 0; j < 4; ++j)
                o[j][r] *= scale;
        }

        // ---- P -> LDS (re-fragment into A layout; wave-private region)
        #pragma unroll
        for (int j = 0; j < 4; ++j)
            #pragma unroll
            for (int r = 0; r < 8; ++r)
                Pst[wave][r + 8 * half][j * 16 + lm] = (__bf16)sc[j][r];

        // ---- O += P V (K dim = keys, 2 steps of 32)
        #pragma unroll
        for (int t = 0; t < 2; ++t) {
            v16bf ap = load_afrag(&Pst[wave][lm][32 * t], half);
            #pragma unroll
            for (int j = 0; j < 4; ++j) {
                v16bf bv;                        // B[key][hd]
                int hd = j * 16 + lm;
                #pragma unroll
                for (int v = 0; v < 8; ++v) {
                    int key = 32 * t + 16 * half + 2 * v;
                    bv[2 * v]     = Vt[key][hd];
                    bv[2 * v + 1] = Vt[key + 1][hd];
                }
                o[j] = __builtin_amdgcn_wmma_f32_16x16x32_bf16(
                    false, ap, false, bv, (short)0, o[j], false, false);
            }
        }
        __syncthreads();   // protect Kt/Vt before next chunk load
    }

    // ---- normalize + store bf16
    #pragma unroll
    for (int j = 0; j < 4; ++j)
        #pragma unroll
        for (int r = 0; r < 8; ++r) {
            int row = qbase + r + 8 * half;
            float val = o[j][r] / rowsum[r];
            Obuf[((size_t)(b * SEQ + row)) * OW + h * DHEAD + j * 16 + lm] = (__bf16)val;
        }
}

// ------------------------------ launcher -----------------------------------
extern "C" void kernel_launch(void* const* d_in, const int* in_sizes, int n_in,
                              void* d_out, int out_size, void* d_ws, size_t ws_size,
                              hipStream_t stream) {
    const float* hs       = (const float*)d_in[0];
    const float* Wq       = (const float*)d_in[1];
    const float* Wk       = (const float*)d_in[2];
    const float* Wv       = (const float*)d_in[3];
    const float* Wo       = (const float*)d_in[4];
    const float* rel_bias = (const float*)d_in[5];
    float* out = (float*)d_out;

    char* ws = (char*)d_ws;
    size_t off = 0;
    __bf16* Wcat   = (__bf16*)(ws + off); off += (size_t)QKVW * DMODEL * 2;   // 1.5 MB
    __bf16* WoEff  = (__bf16*)(ws + off); off += (size_t)DMODEL * OW * 2;     // 0.5 MB
    float*  biasTab= (float*) (ws + off); off += (size_t)KVH * 4096 * 4;      // 64 KB
    __bf16* QKV    = (__bf16*)(ws + off); off += (size_t)MTOT * QKVW * 2;     // 12.6 MB
    __bf16* Obuf   = (__bf16*)(ws + off); off += (size_t)MTOT * OW * 2;       // 4.2 MB
    (void)off; (void)ws_size; (void)in_sizes; (void)n_in; (void)out_size;

    {   // 1) fold weights
        int total = QKVW * DMODEL + DMODEL * OW;
        fold_kernel<<<(total + 255) / 256, 256, 0, stream>>>(Wq, Wk, Wv, Wo, Wcat, WoEff);
    }
    // 2) relative-position bias LUT
    bias_kernel<<<(KVH * 4096 + 255) / 256, 256, 0, stream>>>(rel_bias, biasTab);
    // 3) QKV projection: [8192,1024] x [768,1024]^T -> bf16 [8192,768]
    gemm_f32a_kernel<<<dim3(MTOT / TM, QKVW / TN), 128, 0, stream>>>(
        hs, DMODEL, QKVW, Wcat, QKV);
    // 4) flash attention per (b, kv-head)
    attn_kernel<<<dim3(SEQ / 64, BATCH * KVH), 128, 0, stream>>>(QKV, biasTab, Obuf);
    // 5) output projection: [8192,256] x [1024,256]^T -> fp32 [8192,1024]
    gemm_bf16a_kernel<<<dim3(MTOT / TM, DMODEL / TN), 128, 0, stream>>>(
        Obuf, OW, DMODEL, WoEff, out);
}
